// DecoderRNN_13520557048103
// MI455X (gfx1250) — compile-verified, matrix operand order checked by reading, not run on hardware
//
#include <hip/hip_runtime.h>
#include <hip/hip_bf16.h>
#include <stdint.h>

#define BATCH 32
#define TSTEPS 64
#define VOCAB 50257
#define VOCAB_PAD 50304   // round up to multiple of 128 (block-tile N)
#define EDIM 512
#define HDIM 512
#define G4H 2048          // 4*H
#define MXS 2080          // (T+1)*B rows of xs
#define MXS_PAD 2176      // round up to multiple of 128 (block-tile M)

typedef __attribute__((ext_vector_type(16))) __bf16 bf16x16;
typedef __attribute__((ext_vector_type(8))) float f32x8;

__device__ __forceinline__ unsigned lane_id() { return threadIdx.x & 31u; }

__device__ __forceinline__ unsigned short f2bf(float x) {
  unsigned u = __float_as_uint(x);
  unsigned r = u + 0x7FFFu + ((u >> 16) & 1u);  // round-to-nearest-even
  return (unsigned short)(r >> 16);
}

__device__ __forceinline__ float sigm(float x) { return 1.0f / (1.0f + __expf(-x)); }

// Load a 16x32 bf16 operand fragment from a row-major [rows, 512] matrix
// (K contiguous). `p` points at element (row0, k0). Per-lane layout per CDNA5
// ISA 7.12.2 (16-bit A-matrix 16x32): lanes 0-15 hold row=lane with
// K = {0..7, 16..23}; lanes 16-31 hold row=lane-16 with K = {8..15, 24..31}.
// The B operand (stored as W[N,K] row-major) uses the same pattern, lane = N.
// Two global_load_b128 per lane, no branches.
__device__ __forceinline__ bf16x16 frag_ld(const unsigned short* p) {
  unsigned l = lane_id();
  unsigned r = l & 15u;
  unsigned hi = l >> 4;
  const unsigned short* rp = p + (size_t)r * 512 + hi * 8u;
  union { bf16x16 v; uint4 q[2]; } f;
  f.q[0] = *reinterpret_cast<const uint4*>(rp);        // K-pairs {0..7}
  f.q[1] = *reinterpret_cast<const uint4*>(rp + 16);   // K-pairs {16..23}
  return f.v;
}

#define WMMA_BF16(a, b, c) \
  __builtin_amdgcn_wmma_f32_16x16x32_bf16(false, (a), false, (b), (short)0, (c), false, false)

// Core: acc[2][4] += A[32,512] x { W strip0..3 }^T. Strip g starts at
// Wp + g*strip_elems. Fully unrolled straight-line K loop: no register
// rotation copies (renamer handles it), so the scheduler interleaves each
// chunk's 12 b128 loads with earlier chunks' WMMAs without mov/nop overhead.
__device__ __forceinline__ void mm32(const unsigned short* __restrict__ Ap,
                                     const unsigned short* __restrict__ Wp,
                                     size_t strip_elems, f32x8 acc[2][4]) {
#pragma unroll
  for (int kc = 0; kc < 16; ++kc) {
    const int k0 = kc * 32;
    bf16x16 a0 = frag_ld(Ap + k0);
    bf16x16 a1 = frag_ld(Ap + 16 * 512 + k0);
    bf16x16 w0 = frag_ld(Wp + k0);
    bf16x16 w1 = frag_ld(Wp + strip_elems + k0);
    bf16x16 w2 = frag_ld(Wp + 2 * strip_elems + k0);
    bf16x16 w3 = frag_ld(Wp + 3 * strip_elems + k0);
    acc[0][0] = WMMA_BF16(a0, w0, acc[0][0]);
    acc[1][0] = WMMA_BF16(a1, w0, acc[1][0]);
    acc[0][1] = WMMA_BF16(a0, w1, acc[0][1]);
    acc[1][1] = WMMA_BF16(a1, w1, acc[1][1]);
    acc[0][2] = WMMA_BF16(a0, w2, acc[0][2]);
    acc[1][2] = WMMA_BF16(a1, w2, acc[1][2]);
    acc[0][3] = WMMA_BF16(a0, w3, acc[0][3]);
    acc[1][3] = WMMA_BF16(a1, w3, acc[1][3]);
  }
}

// ---------------------------------------------------------------- prep ----
// Converts f32->bf16 for i < n_src, zero-fills the padding up to n_dst.
__global__ void convert_bf16_kernel(const float* __restrict__ in,
                                    unsigned short* __restrict__ out,
                                    int n_src, int n_dst) {
  for (int i = blockIdx.x * blockDim.x + threadIdx.x; i < n_dst;
       i += gridDim.x * blockDim.x)
    out[i] = (i < n_src) ? f2bf(in[i]) : (unsigned short)0;
}

__global__ void build_xs_kernel(const float* __restrict__ features,
                                const int* __restrict__ captions,
                                const float* __restrict__ embed_w,
                                unsigned short* __restrict__ xs) {
  const int total = MXS_PAD * EDIM;  // includes zero pad rows (t >= 65)
  for (int i = blockIdx.x * blockDim.x + threadIdx.x; i < total;
       i += gridDim.x * blockDim.x) {
    int e = i & (EDIM - 1);
    int tb = i >> 9;           // row m = t*32 + b
    int b = tb & (BATCH - 1);
    int t = tb >> 5;
    float v = 0.0f;
    if (t == 0) {
      v = features[b * EDIM + e];
    } else if (t <= TSTEPS) {
      int tok = captions[b * TSTEPS + (t - 1)];
      v = embed_w[(size_t)tok * EDIM + e];
    }
    xs[i] = f2bf(v);
  }
}

__global__ void zero_kernel(float* __restrict__ p, int n) {
  for (int i = blockIdx.x * blockDim.x + threadIdx.x; i < n;
       i += gridDim.x * blockDim.x)
    p[i] = 0.0f;
}

// ------------------------------------------------------- generic GEMM ----
// C[., ldc] = A_pad @ B_pad^T (+ bias). Operands padded so every fragment
// load is in-bounds; only the store clips columns at Nclip. Block tile
// 128x128, 8 waves, each wave 32(M) x 64(N) via 2x4 accumulator tiles.
__global__ __launch_bounds__(256) void gemm_bf16_kernel(
    const unsigned short* __restrict__ A,   // [M_pad, 512] bf16 row-major
    const unsigned short* __restrict__ Bm,  // [N_pad, 512] bf16 row-major
    const float* __restrict__ bias,         // [Nclip] or nullptr
    float* __restrict__ C,                  // [M, ldc] f32
    int Nclip, int ldc) {
  const int w = threadIdx.x >> 5;
  const int m0 = blockIdx.y * 128 + (w & 3) * 32;
  const int n0 = blockIdx.x * 128 + (w >> 2) * 64;
  f32x8 acc[2][4] = {};
  mm32(A + (size_t)m0 * 512, Bm + (size_t)n0 * 512, (size_t)16 * 512, acc);

  const unsigned l = lane_id();
  const int nloc = (int)(l & 15u);
  const int hi = (int)(l >> 4);
#pragma unroll
  for (int nt = 0; nt < 4; ++nt) {
    const int n = n0 + nt * 16 + nloc;
    if (n < Nclip) {
      const float bv = (bias != nullptr) ? bias[n] : 0.0f;
#pragma unroll
      for (int mt = 0; mt < 2; ++mt) {
#pragma unroll
        for (int r = 0; r < 8; ++r) {
          const int m = m0 + mt * 16 + r + hi * 8;
          C[(size_t)m * ldc + n] = acc[mt][nt][r] + bv;
        }
      }
    }
  }
}

// -------------------------------------------------------- LSTM step ----
// gates[32,2048] = (gx or x@w_ih^T) + h_in@w_hh^T + bias, then cell update.
// 8 blocks x 4 waves = 32 waves; each wave owns a 16-wide H column strip and
// computes all four gates (i,f,g,o) for it, so the nonlinearity stays local.
__global__ __launch_bounds__(128) void lstm_step_kernel(
    const unsigned short* __restrict__ x,     // [32,512] bf16 (unused if gx)
    const unsigned short* __restrict__ h_in,  // [32,512] bf16
    const unsigned short* __restrict__ w_ih,  // [2048,512] bf16 (unused if gx)
    const unsigned short* __restrict__ w_hh,  // [2048,512] bf16
    const float* __restrict__ gx,             // [32,2048] precomputed x-gates or null
    const float* __restrict__ bias,           // [2048]
    float* __restrict__ c,                    // [32,512] f32 cell state (in/out)
    unsigned short* __restrict__ h_out,       // [32,512] bf16
    unsigned short* __restrict__ hs_out,      // optional extra copy of h
    int hs_stride) {
  const int wave = blockIdx.x * (blockDim.x >> 5) + (threadIdx.x >> 5);  // 0..31
  const int col0 = wave * 16;
  const unsigned l = lane_id();
  const int nloc = (int)(l & 15u);
  const int hi = (int)(l >> 4);
  const int col = col0 + nloc;

  f32x8 acc[2][4] = {};
  if (gx != nullptr) {
#pragma unroll
    for (int mt = 0; mt < 2; ++mt)
#pragma unroll
      for (int g = 0; g < 4; ++g)
#pragma unroll
        for (int r = 0; r < 8; ++r) {
          const int m = mt * 16 + r + hi * 8;
          acc[mt][g][r] = gx[(size_t)m * G4H + g * HDIM + col];
        }
  } else {
    mm32(x, w_ih + (size_t)col0 * 512, (size_t)512 * 512, acc);
  }
  // recurrent contribution: h_in @ w_hh^T
  mm32(h_in, w_hh + (size_t)col0 * 512, (size_t)512 * 512, acc);

  const float bi = bias[col];
  const float bfv = bias[HDIM + col];
  const float bg = bias[2 * HDIM + col];
  const float bo = bias[3 * HDIM + col];
#pragma unroll
  for (int mt = 0; mt < 2; ++mt) {
#pragma unroll
    for (int r = 0; r < 8; ++r) {
      const int m = mt * 16 + r + hi * 8;
      const size_t idx = (size_t)m * HDIM + col;
      const float iv = acc[mt][0][r] + bi;
      const float fv = acc[mt][1][r] + bfv;
      const float gv = acc[mt][2][r] + bg;
      const float ov = acc[mt][3][r] + bo;
      const float cn = sigm(fv) * c[idx] + sigm(iv) * tanhf(gv);
      const float hn = sigm(ov) * tanhf(cn);
      c[idx] = cn;
      const unsigned short hb = f2bf(hn);
      h_out[idx] = hb;
      if (hs_out != nullptr) hs_out[(size_t)m * hs_stride + col] = hb;
    }
  }
}

// ---------------------------------------------------------------- host ----
extern "C" void kernel_launch(void* const* d_in, const int* in_sizes, int n_in,
                              void* d_out, int out_size, void* d_ws, size_t ws_size,
                              hipStream_t stream) {
  (void)in_sizes; (void)n_in; (void)out_size; (void)ws_size;
  const float* features = (const float*)d_in[0];
  const int*   captions = (const int*)d_in[1];
  const float* embed_w  = (const float*)d_in[2];
  const float* w_ih0    = (const float*)d_in[3];
  const float* w_hh0    = (const float*)d_in[4];
  const float* b0       = (const float*)d_in[5];
  const float* w_ih1    = (const float*)d_in[6];
  const float* w_hh1    = (const float*)d_in[7];
  const float* b1       = (const float*)d_in[8];
  const float* w_out    = (const float*)d_in[9];
  const float* b_out    = (const float*)d_in[10];
  float* out = (float*)d_out;

  char* ws = (char*)d_ws;
  size_t off = 0;
  auto take = [&](size_t bytes) -> char* {
    char* p = ws + off;
    off += (bytes + 255) & ~(size_t)255;
    return p;
  };
  unsigned short* xs    = (unsigned short*)take((size_t)MXS_PAD * EDIM * 2);       // padded rows
  unsigned short* wih0b = (unsigned short*)take((size_t)G4H * EDIM * 2);
  unsigned short* whh0b = (unsigned short*)take((size_t)G4H * HDIM * 2);
  unsigned short* wih1b = (unsigned short*)take((size_t)G4H * HDIM * 2);
  unsigned short* whh1b = (unsigned short*)take((size_t)G4H * HDIM * 2);
  unsigned short* woutb = (unsigned short*)take((size_t)VOCAB_PAD * HDIM * 2);     // padded rows
  unsigned short* hsbuf = (unsigned short*)take((size_t)BATCH * TSTEPS * HDIM * 2);// [b][t][h]
  float* gx0            = (float*)take((size_t)MXS_PAD * G4H * 4);                 // padded rows
  float* state          = (float*)take(4 * (size_t)BATCH * HDIM * 2 + 2 * (size_t)BATCH * HDIM * 4);
  unsigned short* h0a = (unsigned short*)state;
  unsigned short* h0b = h0a + BATCH * HDIM;
  unsigned short* h1a = h0b + BATCH * HDIM;
  unsigned short* h1b = h1a + BATCH * HDIM;
  float* c0 = (float*)(h1b + BATCH * HDIM);
  float* c1 = c0 + BATCH * HDIM;

  // 1) bf16 conversions (w_out zero-padded to 50304 rows) + inputs + state
  convert_bf16_kernel<<<512, 256, 0, stream>>>(w_ih0, wih0b, G4H * EDIM, G4H * EDIM);
  convert_bf16_kernel<<<512, 256, 0, stream>>>(w_hh0, whh0b, G4H * HDIM, G4H * HDIM);
  convert_bf16_kernel<<<512, 256, 0, stream>>>(w_ih1, wih1b, G4H * HDIM, G4H * HDIM);
  convert_bf16_kernel<<<512, 256, 0, stream>>>(w_hh1, whh1b, G4H * HDIM, G4H * HDIM);
  convert_bf16_kernel<<<4096, 256, 0, stream>>>(w_out, woutb, VOCAB * HDIM,
                                                VOCAB_PAD * HDIM);
  build_xs_kernel<<<512, 256, 0, stream>>>(features, captions, embed_w, xs);
  zero_kernel<<<64, 256, 0, stream>>>(state, 4 * BATCH * HDIM);  // h0a/b,h1a/b,c0,c1

  // 2) hoisted recurrence-independent GEMM: gx0[t*32+b, :] = xs @ w_ih0^T
  {
    dim3 g(G4H / 128, MXS_PAD / 128);  // (16, 17); pad rows produce junk-free zeros
    gemm_bf16_kernel<<<g, 256, 0, stream>>>(xs, wih0b, nullptr, gx0, G4H, G4H);
  }

  // 3) sequential recurrence: 65 steps x 2 layers, ping-pong h buffers
  for (int t = 0; t <= TSTEPS; ++t) {
    unsigned short* h0_in  = (t & 1) ? h0b : h0a;
    unsigned short* h0_out = (t & 1) ? h0a : h0b;
    unsigned short* h1_in  = (t & 1) ? h1b : h1a;
    unsigned short* h1_out = (t & 1) ? h1a : h1b;
    lstm_step_kernel<<<8, 128, 0, stream>>>(
        nullptr, h0_in, nullptr, whh0b,
        gx0 + (size_t)t * BATCH * G4H, b0, c0, h0_out, nullptr, 0);
    unsigned short* hsp =
        (t >= 1) ? hsbuf + (size_t)(t - 1) * HDIM : nullptr;  // row b*T+(t-1)
    lstm_step_kernel<<<8, 128, 0, stream>>>(
        h0_out, h1_in, wih1b, whh1b, nullptr, b1, c1, h1_out,
        hsp, TSTEPS * HDIM);
  }

  // 4) output projection: logits[b*T+t, v] = hs @ w_out^T + b_out
  {
    dim3 g(VOCAB_PAD / 128, (BATCH * TSTEPS) / 128);  // (393, 16)
    gemm_bf16_kernel<<<g, 256, 0, stream>>>(hsbuf, woutb, b_out, out,
                                            VOCAB, VOCAB);
  }
}
